// FramePrimerDecoder_39659728011316
// MI455X (gfx1250) — compile-verified, matrix operand order checked by reading, not run on hardware
//
#include <hip/hip_runtime.h>
#include <hip/hip_bf16.h>

// ---------------- types ----------------
typedef __attribute__((ext_vector_type(16))) __bf16 v16bf;
typedef __attribute__((ext_vector_type(8)))  __bf16 v8bf;
typedef __attribute__((ext_vector_type(4)))  __bf16 v4bf;
typedef __attribute__((ext_vector_type(8)))  float  v8f;
typedef __attribute__((ext_vector_type(4)))  unsigned v4u;
typedef __attribute__((ext_vector_type(8)))  int v8i;
typedef __attribute__((ext_vector_type(4)))  int v4i;

#define WDIM 1024
#define BINS 1024
#define BB   4
#define DHEAD 256
#define FFD  2048
#define MEG  1048576L

#if defined(__has_builtin)
#  if __has_builtin(__builtin_amdgcn_tensor_load_to_lds) && __has_builtin(__builtin_amdgcn_s_wait_tensorcnt)
#    define HAS_TDM 1
#  endif
#endif
#ifndef HAS_TDM
#  define HAS_TDM 0
#endif

__device__ __forceinline__ __bf16 f2bf(float f) {
    unsigned u = __builtin_bit_cast(unsigned, f);
    unsigned r = u + 0x7FFFu + ((u >> 16) & 1u);
    unsigned short s = (unsigned short)(r >> 16);
    return __builtin_bit_cast(__bf16, s);
}

__device__ __forceinline__ v16bf load_frag(const __bf16* base, int row, int cb) {
    const v8bf* p0 = (const v8bf*)(base + row * 40 + cb);
    const v8bf* p1 = (const v8bf*)(base + row * 40 + cb + 16);
    v8bf lo = *p0, hi = *p1;
    v16bf r;
#pragma unroll
    for (int i = 0; i < 8; ++i) { r[i] = lo[i]; r[i + 8] = hi[i]; }
    return r;
}

// Issue a TDM load of a 128-row x 32-col bf16 tile (row stride = strideElems)
// into LDS with 16B padding every 64B -> LDS row pitch 40 bf16 (conflict-free).
__device__ __forceinline__ void tdm_load_tile(const __bf16* gsrc, int strideElems, __bf16* ldst) {
#if HAS_TDM
    unsigned long long ga = (unsigned long long)gsrc;
    v4u g0;
    g0[0] = 1u;                                            // count=1, user mode
    g0[1] = (unsigned)(unsigned long long)ldst;            // lds byte address
    g0[2] = (unsigned)(ga & 0xFFFFFFFFull);                // global_addr[31:0]
    g0[3] = (unsigned)((ga >> 32) & 0x01FFFFFFull) | 0x80000000u;  // addr[56:32] | type=2
    v8i g1;
    g1[0] = (int)((1u << 16) |        // data_size = 2 bytes
                  (1u << 20) |        // pad_enable
                  (3u << 22) |        // pad_interval: 8<<3 = 64B
                  (3u << 25));        // pad_amount: 4 DWORDs = 16B
    g1[1] = (int)(32u << 16);         // tensor_dim0 = 32
    g1[2] = (int)(128u << 16);        // tensor_dim1 = 128 (low16)
    g1[3] = (int)(32u << 16);         // tile_dim0 = 32
    g1[4] = 128;                      // tile_dim1 = 128, tile_dim2 = 0
    g1[5] = (int)(unsigned)strideElems; // tensor_dim0_stride[31:0]
    g1[6] = 0;
    g1[7] = 0;
    v4i gz4 = {0, 0, 0, 0};
    v8i gz8 = {0, 0, 0, 0, 0, 0, 0, 0};
    __builtin_amdgcn_tensor_load_to_lds(g0, g1, gz4, gz4, gz8, 0);
#else
    (void)gsrc; (void)strideElems; (void)ldst;
#endif
}

template <short N>
__device__ __forceinline__ void tdm_wait() {
#if HAS_TDM
    __builtin_amdgcn_s_wait_tensorcnt(N);
#endif
}

// fallback manual staging: 128x32 bf16 row-major tile -> LDS pitch 40
__device__ __forceinline__ void stage_tile(const __bf16* g, int ld, __bf16* ldst, int tid) {
#pragma unroll
    for (int j = 0; j < 2; ++j) {
        int ch = tid + j * 256;           // 512 chunks of 8 elems
        int row = ch >> 2;                // 4 chunks per row
        int cc = (ch & 3) << 3;
        v8bf v = *(const v8bf*)(g + (long)row * ld + cc);
        *(v8bf*)(ldst + row * 40 + cc) = v;
    }
}

// staging with transpose: source [K,N] bf16 (ld = N row stride) -> LDS [n][k] pitch 40
__device__ __forceinline__ void stage_tile_bt(const __bf16* g, int ld, __bf16* ldst, int tid) {
#pragma unroll
    for (int j = 0; j < 2; ++j) {
        int ch = tid + j * 256;
        int kr = ch >> 4;                 // 16 chunks per k-row
        int nn = (ch & 15) << 3;
        v8bf v = *(const v8bf*)(g + (long)kr * ld + nn);
#pragma unroll
        for (int i = 0; i < 8; ++i) ldst[(nn + i) * 40 + kr] = v[i];
    }
}

// ---------------- WMMA GEMM ----------------
// C[M,N] = A[M,K] (bf16 row major) x B^T
//   BT==false : B is [N,K] bf16 row major      (A and B tiles DMA'd by TDM)
//   BT==true  : B is [K,N] bf16 row major      (A via TDM, B manual transpose staging)
// Epilogues:
//   0: C(f32) = acc
//   1: C(f32) = (acc + X1[col*ldx1+row]) * scale + X2[row*ldx2+col]   (qk; skew == transpose)
//   3: C(f32) = acc + X1[row*ldx1+col]                                 (residual)
//   4: C(bf16) = acc
//   5: C(bf16) = relu(acc)^2
template <int EPI, bool BT>
__global__ __launch_bounds__(256, 2) void gemm_bf16_wmma(
    const __bf16* __restrict__ A, int lda, long sAb, long sAn,
    const __bf16* __restrict__ Bm, int ldb, long sBb, long sBn,
    float* __restrict__ C, int ldc, long sCb, long sCn,
    const float* __restrict__ X1, int ldx1, long sX1b, long sX1n,
    const float* __restrict__ X2, int ldx2, long sX2b, long sX2n,
    int M, int N, int K, float scale)
{
    __shared__ __bf16 lds[2][2][128 * 40];

    const int z = blockIdx.z, zb = z >> 2, zn = z & 3;
    A  += zb * sAb + zn * sAn;
    Bm += zb * sBb + zn * sBn;
    long coff = zb * sCb + zn * sCn;
    if (X1) X1 += zb * sX1b + zn * sX1n;
    if (X2) X2 += zb * sX2b + zn * sX2n;

    const int m0 = blockIdx.y * 128, n0 = blockIdx.x * 128;
    const int tid = threadIdx.x;
    const int wid = tid >> 5, lane = tid & 31;
    const int wrow = (wid >> 1) * 32, wcol = (wid & 1) * 64;
    const int lrow = lane & 15, sel = lane >> 4;
    const int cb = sel ? 8 : 0;
    const bool loader = (wid == 0);

    const __bf16* Abase  = A + (long)m0 * lda;
    const __bf16* Bbase  = BT ? (Bm + n0) : (Bm + (long)n0 * ldb);

    v8f acc[2][4] = {};
    const int kt_total = K >> 5;

    // preload tile 0
#if HAS_TDM
    if (loader) {
        tdm_load_tile(Abase, lda, &lds[0][0][0]);
        if (!BT) tdm_load_tile(Bbase, ldb, &lds[0][1][0]);
    }
#else
    stage_tile(Abase, lda, &lds[0][0][0], tid);
    if (!BT) stage_tile(Bbase, ldb, &lds[0][1][0], tid);
#endif
    if (BT) stage_tile_bt(Bbase, ldb, &lds[0][1][0], tid);

    for (int kt = 0; kt < kt_total; ++kt) {
        const int cur = kt & 1, nxt = cur ^ 1;
        __syncthreads();   // everyone finished reading the `nxt` buffer (iter kt-1)
        if (kt + 1 < kt_total) {
            const int kn = (kt + 1) << 5;
#if HAS_TDM
            if (loader) {
                tdm_load_tile(Abase + kn, lda, &lds[nxt][0][0]);
                if (!BT) tdm_load_tile(Bbase + kn, ldb, &lds[nxt][1][0]);
            }
#else
            stage_tile(Abase + kn, lda, &lds[nxt][0][0], tid);
            if (!BT) stage_tile(Bbase + kn, ldb, &lds[nxt][1][0], tid);
#endif
            if (BT) stage_tile_bt(Bbase + (long)kn * ldb, ldb, &lds[nxt][1][0], tid);
            if (loader) tdm_wait<BT ? (short)1 : (short)2>();  // tile kt complete (in-order)
        } else {
            if (loader) tdm_wait<(short)0>();
        }
        __syncthreads();   // tile `cur` visible to all waves

        const __bf16* sA = &lds[cur][0][0];
        const __bf16* sB = &lds[cur][1][0];
        v16bf af[2], bfv[4];
        af[0] = load_frag(sA, wrow + lrow, cb);
        af[1] = load_frag(sA, wrow + 16 + lrow, cb);
#pragma unroll
        for (int nt = 0; nt < 4; ++nt)
            bfv[nt] = load_frag(sB, wcol + nt * 16 + lrow, cb);
#pragma unroll
        for (int mt = 0; mt < 2; ++mt)
#pragma unroll
            for (int nt = 0; nt < 4; ++nt)
                acc[mt][nt] = __builtin_amdgcn_wmma_f32_16x16x32_bf16(
                    false, af[mt], false, bfv[nt], (short)0, acc[mt][nt], false, false);
    }

    // ---- epilogue ----
    float* Cf = C + coff;
    __bf16* Cb = (__bf16*)C + coff;
#pragma unroll
    for (int mt = 0; mt < 2; ++mt)
#pragma unroll
        for (int nt = 0; nt < 4; ++nt)
#pragma unroll
            for (int r = 0; r < 8; ++r) {
                int row = m0 + wrow + mt * 16 + (sel ? r + 8 : r);
                int col = n0 + wcol + nt * 16 + lrow;
                float v = acc[mt][nt][r];
                long cidx = (long)row * ldc + col;
                if (EPI == 0) {
                    Cf[cidx] = v;
                } else if (EPI == 1) {
                    float t = v + X1[(long)col * ldx1 + row];
                    Cf[cidx] = t * scale + X2[(long)row * ldx2 + col];
                } else if (EPI == 3) {
                    Cf[cidx] = v + X1[(long)row * ldx1 + col];
                } else if (EPI == 4) {
                    Cb[cidx] = f2bf(v);
                } else { // EPI == 5
                    float t = v > 0.f ? v : 0.f;
                    Cb[cidx] = f2bf(t * t);
                }
            }
}

// ---------------- input projection: out[b,w,h] = sum_c x[b,c,h,w]*w[c] + bias ----------------
template <bool BF>
__global__ __launch_bounds__(256) void inproj_kernel(
    const float* __restrict__ x, const float* __restrict__ wc,
    const float* __restrict__ bptr, void* __restrict__ outv)
{
    __shared__ float tile[32][33];
    const int b = blockIdx.z, h0 = blockIdx.y * 32, w0 = blockIdx.x * 32;
    const int tx = threadIdx.x, ty = threadIdx.y;
    const float bias = bptr[0];
    float wv[8];
#pragma unroll
    for (int c = 0; c < 8; ++c) wv[c] = wc[c];
    for (int hh = ty; hh < 32; hh += 8) {
        float s = 0.f;
#pragma unroll
        for (int c = 0; c < 8; ++c)
            s += x[(((long)b * 8 + c) * BINS + h0 + hh) * WDIM + w0 + tx] * wv[c];
        tile[hh][tx] = s + bias;
    }
    __syncthreads();
    for (int ww = ty; ww < 32; ww += 8) {
        long o = ((long)b * WDIM + w0 + ww) * BINS + h0 + tx;
        if (BF) ((__bf16*)outv)[o] = f2bf(tile[tx][ww]);
        else    ((float*)outv)[o] = tile[tx][ww];
    }
}

// ---------------- LayerNorm over last dim (1024), bf16 output ----------------
__global__ __launch_bounds__(256) void ln_kernel(
    const float* __restrict__ in, const float* __restrict__ g,
    const float* __restrict__ bb, __bf16* __restrict__ out)
{
    __shared__ float red[256];
    const int tid = threadIdx.x;
    const long base = (long)blockIdx.x * BINS;
    float s1 = 0.f, s2 = 0.f;
#pragma unroll
    for (int j = 0; j < 4; ++j) {
        float v = in[base + tid + j * 256];
        s1 += v; s2 += v * v;
    }
    red[tid] = s1; __syncthreads();
    for (int s = 128; s > 0; s >>= 1) { if (tid < s) red[tid] += red[tid + s]; __syncthreads(); }
    float mean = red[0] * (1.f / BINS); __syncthreads();
    red[tid] = s2; __syncthreads();
    for (int s = 128; s > 0; s >>= 1) { if (tid < s) red[tid] += red[tid + s]; __syncthreads(); }
    float var = red[0] * (1.f / BINS) - mean * mean;
    float rs = rsqrtf(var + 1e-5f);
#pragma unroll
    for (int j = 0; j < 4; ++j) {
        int c = tid + j * 256;
        out[base + c] = f2bf((in[base + c] - mean) * rs * g[c] + bb[c]);
    }
}

// ---------------- depthwise conv1d k=3 pad=1 along t; [b,t,c] f32 -> bf16 ----------------
__global__ __launch_bounds__(256) void dwconv_kernel(
    const float* __restrict__ in, const float* __restrict__ w3, __bf16* __restrict__ out)
{
    long gid = (long)blockIdx.x * 256 + threadIdx.x;
    int c = (int)(gid % 256) * 4;
    long bt = gid / 256;
    int t = (int)(bt % WDIM);
    int b = (int)(bt / WDIM);
    long base = ((long)b * WDIM + t) * BINS + c;
    float4 x0 = *(const float4*)(in + base);
    float4 xm = make_float4(0.f, 0.f, 0.f, 0.f), xp = make_float4(0.f, 0.f, 0.f, 0.f);
    if (t > 0)        xm = *(const float4*)(in + base - BINS);
    if (t < WDIM - 1) xp = *(const float4*)(in + base + BINS);
    v4bf o;
    o[0] = f2bf(xm.x * w3[(c+0)*3] + x0.x * w3[(c+0)*3+1] + xp.x * w3[(c+0)*3+2]);
    o[1] = f2bf(xm.y * w3[(c+1)*3] + x0.y * w3[(c+1)*3+1] + xp.y * w3[(c+1)*3+2]);
    o[2] = f2bf(xm.z * w3[(c+2)*3] + x0.z * w3[(c+2)*3+1] + xp.z * w3[(c+2)*3+2]);
    o[3] = f2bf(xm.w * w3[(c+3)*3] + x0.w * w3[(c+3)*3+1] + xp.w * w3[(c+3)*3+2]);
    *(v4bf*)(out + base) = o;
}

// ---------------- row softmax (len 1024), f32 in -> bf16 out ----------------
__global__ __launch_bounds__(256) void softmax_kernel(
    const float* __restrict__ in, __bf16* __restrict__ out)
{
    __shared__ float red[256];
    const int tid = threadIdx.x;
    const long base = (long)blockIdx.x * WDIM;
    float x[4];
    float mx = -3.4e38f;
#pragma unroll
    for (int j = 0; j < 4; ++j) { x[j] = in[base + tid + j * 256]; mx = fmaxf(mx, x[j]); }
    red[tid] = mx; __syncthreads();
    for (int s = 128; s > 0; s >>= 1) { if (tid < s) red[tid] = fmaxf(red[tid], red[tid + s]); __syncthreads(); }
    mx = red[0]; __syncthreads();
    float sm = 0.f;
#pragma unroll
    for (int j = 0; j < 4; ++j) { x[j] = __expf(x[j] - mx); sm += x[j]; }
    red[tid] = sm; __syncthreads();
    for (int s = 128; s > 0; s >>= 1) { if (tid < s) red[tid] += red[tid + s]; __syncthreads(); }
    float inv = 1.f / red[0];
#pragma unroll
    for (int j = 0; j < 4; ++j) out[base + tid + j * 256] = f2bf(x[j] * inv);
}

// ---------------- f32 -> bf16 bulk convert ----------------
__global__ __launch_bounds__(256) void cvt_bf16_kernel(
    const float* __restrict__ in, __bf16* __restrict__ out)
{
    long i = ((long)blockIdx.x * 256 + threadIdx.x) * 4;
    float4 v = *(const float4*)(in + i);
    v4bf o; o[0] = f2bf(v.x); o[1] = f2bf(v.y); o[2] = f2bf(v.z); o[3] = f2bf(v.w);
    *(v4bf*)(out + i) = o;
}

// ---------------- 32x32 tiled transpose: out[c, r] = in[r, c] ----------------
template <bool BF_OUT>
__global__ __launch_bounds__(256) void transpose32_kernel(
    const float* __restrict__ in, long in_z, int ld_in,
    void* __restrict__ outv, long out_z, int ld_out)
{
    __shared__ float tile[32][33];
    in += (long)blockIdx.z * in_z;
    const int r0 = blockIdx.y * 32, c0 = blockIdx.x * 32;
    const int tx = threadIdx.x, ty = threadIdx.y;
    for (int rr = ty; rr < 32; rr += 8)
        tile[rr][tx] = in[(long)(r0 + rr) * ld_in + c0 + tx];
    __syncthreads();
    for (int cc = ty; cc < 32; cc += 8) {
        long o = (long)blockIdx.z * out_z + (long)(c0 + cc) * ld_out + r0 + tx;
        if (BF_OUT) ((__bf16*)outv)[o] = f2bf(tile[tx][cc]);
        else        ((float*)outv)[o] = tile[tx][cc];
    }
}

// ---------------- host-side GEMM dispatcher ----------------
static void launch_gemm(hipStream_t s, int epi, bool bt, int M, int N, int Z,
                        const __bf16* A, int lda, long sAb, long sAn,
                        const __bf16* B, int ldb, long sBb, long sBn,
                        void* C, int ldc, long sCb, long sCn,
                        const float* X1, int ldx1, long sX1b, long sX1n,
                        const float* X2, int ldx2, long sX2b, long sX2n,
                        int K, float scale)
{
    dim3 grid(N / 128, M / 128, Z);
#define GEMM_GO(E, T) gemm_bf16_wmma<E, T><<<grid, 256, 0, s>>>( \
        A, lda, sAb, sAn, B, ldb, sBb, sBn, (float*)C, ldc, sCb, sCn, \
        X1, ldx1, sX1b, sX1n, X2, ldx2, sX2b, sX2n, M, N, K, scale)
    if (bt) { GEMM_GO(4, true); return; }
    switch (epi) {
        case 0: GEMM_GO(0, false); break;
        case 1: GEMM_GO(1, false); break;
        case 3: GEMM_GO(3, false); break;
        case 5: GEMM_GO(5, false); break;
    }
#undef GEMM_GO
}

extern "C" void kernel_launch(void* const* d_in, const int* in_sizes, int n_in,
                              void* d_out, int out_size, void* d_ws, size_t ws_size,
                              hipStream_t stream)
{
    (void)in_sizes; (void)n_in; (void)out_size; (void)ws_size;
    const float* x     = (const float*)d_in[0];
    const float* mem   = (const float*)d_in[1];
    const float* prevqk[2] = {(const float*)d_in[2], (const float*)d_in[3]};
    const float* ip_w  = (const float*)d_in[4];
    const float* ip_b  = (const float*)d_in[5];
    const float* mp_w  = (const float*)d_in[6];
    const float* mp_b  = (const float*)d_in[7];
    const float* ln_g[3] = {(const float*)d_in[8],  (const float*)d_in[10], (const float*)d_in[12]};
    const float* ln_b[3] = {(const float*)d_in[9],  (const float*)d_in[11], (const float*)d_in[13]};
    const float* a_qw[2] = {(const float*)d_in[14], (const float*)d_in[22]};
    const float* a_qc[2] = {(const float*)d_in[15], (const float*)d_in[23]};
    const float* a_kw[2] = {(const float*)d_in[16], (const float*)d_in[24]};
    const float* a_kc[2] = {(const float*)d_in[17], (const float*)d_in[25]};
    const float* a_vw[2] = {(const float*)d_in[18], (const float*)d_in[26]};
    const float* a_vc[2] = {(const float*)d_in[19], (const float*)d_in[27]};
    const float* a_ow[2] = {(const float*)d_in[20], (const float*)d_in[28]};
    const float* a_pw[2] = {(const float*)d_in[21], (const float*)d_in[29]};
    const float* l1_w  = (const float*)d_in[30];
    const float* l2_w  = (const float*)d_in[31];

    float* outp = (float*)d_out;
    float* qk_out[2] = {outp + 4 * MEG, outp + 20 * MEG};

    char* wsb = (char*)d_ws;
    const long MB = 1 << 20;
    float*  XS   = (float*)(wsb + 0);          // [b,w,h] f32, 16MB
    float*  TMP  = (float*)(wsb + 16 * MB);    // proj pre-conv f32, 16MB
    float*  PB   = (float*)(wsb + 32 * MB);    // P bias f32, 64MB
    __bf16* LNB  = (__bf16*)(wsb + 96 * MB);   // 8MB
    __bf16* MSB  = (__bf16*)(wsb + 104 * MB);  // 8MB
    __bf16* QB   = (__bf16*)(wsb + 112 * MB);  // 8MB
    __bf16* KB   = (__bf16*)(wsb + 120 * MB);  // 8MB
    __bf16* VB   = (__bf16*)(wsb + 128 * MB);  // 8MB
    __bf16* PR   = (__bf16*)(wsb + 136 * MB);  // probs, 32MB
    __bf16* TMPB = (__bf16*)(wsb + 168 * MB);  // attn out bf16, 8MB
    __bf16* FFHB = (__bf16*)(wsb + 176 * MB);  // ffn hidden bf16, 16MB
    __bf16* WQB  = (__bf16*)(wsb + 192 * MB);  // 2MB each
    __bf16* WKB  = (__bf16*)(wsb + 194 * MB);
    __bf16* WVB  = (__bf16*)(wsb + 196 * MB);
    __bf16* WOB  = (__bf16*)(wsb + 198 * MB);
    __bf16* L1B  = (__bf16*)(wsb + 200 * MB);  // 4MB
    __bf16* L2B  = (__bf16*)(wsb + 204 * MB);  // 4MB
    __bf16* PWTB = (__bf16*)(wsb + 208 * MB);  // 0.5MB

    const dim3 tb(32, 8);
    const dim3 tgrid(WDIM / 32, BINS / 32, BB);
    const int MROWS = BB * WDIM;
    const long sBat = (long)WDIM * BINS;       // per-batch activation stride (elems)
    const long sQKn = (long)WDIM * WDIM;
    const long sQKb = 4 * sQKn;
    const float qscale = 1.0f / 32.0f;         // 1/sqrt(BINS)

    // input projections
    inproj_kernel<false><<<tgrid, tb, 0, stream>>>(x,   ip_w, ip_b, XS);
    inproj_kernel<true ><<<tgrid, tb, 0, stream>>>(mem, mp_w, mp_b, MSB);
    // FFN weights -> bf16 (once)
    cvt_bf16_kernel<<<2048, 256, 0, stream>>>(l1_w, L1B);
    cvt_bf16_kernel<<<2048, 256, 0, stream>>>(l2_w, L2B);

    for (int att = 0; att < 2; ++att) {
        const __bf16* kv_src = (att == 0) ? LNB : MSB;
        // weights -> bf16
        cvt_bf16_kernel<<<1024, 256, 0, stream>>>(a_qw[att], WQB);
        cvt_bf16_kernel<<<1024, 256, 0, stream>>>(a_kw[att], WKB);
        cvt_bf16_kernel<<<1024, 256, 0, stream>>>(a_vw[att], WVB);
        cvt_bf16_kernel<<<1024, 256, 0, stream>>>(a_ow[att], WOB);
        // pw [d, W] -> PWTB [W, d] bf16
        transpose32_kernel<true><<<dim3(WDIM / 32, DHEAD / 32, 1), tb, 0, stream>>>(
            a_pw[att], 0, WDIM, PWTB, 0, DHEAD);

        ln_kernel<<<MROWS, 256, 0, stream>>>(XS, ln_g[att], ln_b[att], LNB);

        // Q/K/V projections + dwconv -> bf16
        launch_gemm(stream, 0, false, MROWS, BINS, 1, LNB, BINS, 0, 0, WQB, BINS, 0, 0,
                    TMP, BINS, 0, 0, nullptr, 0, 0, 0, nullptr, 0, 0, 0, BINS, 1.f);
        dwconv_kernel<<<4096, 256, 0, stream>>>(TMP, a_qc[att], QB);
        launch_gemm(stream, 0, false, MROWS, BINS, 1, kv_src, BINS, 0, 0, WKB, BINS, 0, 0,
                    TMP, BINS, 0, 0, nullptr, 0, 0, 0, nullptr, 0, 0, 0, BINS, 1.f);
        dwconv_kernel<<<4096, 256, 0, stream>>>(TMP, a_kc[att], KB);
        launch_gemm(stream, 0, false, MROWS, BINS, 1, kv_src, BINS, 0, 0, WVB, BINS, 0, 0,
                    TMP, BINS, 0, 0, nullptr, 0, 0, 0, nullptr, 0, 0, 0, BINS, 1.f);
        dwconv_kernel<<<4096, 256, 0, stream>>>(TMP, a_vc[att], VB);

        // P[b,n] = q @ pw  -> PB (f32)
        launch_gemm(stream, 0, false, WDIM, WDIM, BB * 4,
                    QB, BINS, sBat, DHEAD, PWTB, DHEAD, 0, 0,
                    PB, WDIM, sQKb, sQKn,
                    nullptr, 0, 0, 0, nullptr, 0, 0, 0, DHEAD, 1.f);
        // scores: qk = (q.k^T + P^T)/sqrt(c) + prev -> d_out
        launch_gemm(stream, 1, false, WDIM, WDIM, BB * 4,
                    QB, BINS, sBat, DHEAD, KB, BINS, sBat, DHEAD,
                    qk_out[att], WDIM, sQKb, sQKn,
                    PB, WDIM, sQKb, sQKn,
                    prevqk[att], WDIM, sQKb, sQKn, DHEAD, qscale);
        // softmax -> bf16 probs
        softmax_kernel<<<BB * 4 * WDIM, 256, 0, stream>>>(qk_out[att], PR);
        // attn = probs @ V  (B is [K,N]) -> bf16 band columns of TMPB
        launch_gemm(stream, 4, true, WDIM, DHEAD, BB * 4,
                    PR, WDIM, sQKb, sQKn, VB, BINS, sBat, DHEAD,
                    TMPB, BINS, sBat, DHEAD,
                    nullptr, 0, 0, 0, nullptr, 0, 0, 0, WDIM, 1.f);
        // output projection + residual: xs += attn @ ow^T
        launch_gemm(stream, 3, false, MROWS, BINS, 1, TMPB, BINS, 0, 0, WOB, BINS, 0, 0,
                    XS, BINS, 0, 0, XS, BINS, 0, 0, nullptr, 0, 0, 0, BINS, 1.f);
    }

    // FFN: xs += relu(ln3(xs) @ l1^T)^2 @ l2^T
    ln_kernel<<<MROWS, 256, 0, stream>>>(XS, ln_g[2], ln_b[2], LNB);
    launch_gemm(stream, 5, false, MROWS, FFD, 1, LNB, BINS, 0, 0, L1B, BINS, 0, 0,
                FFHB, FFD, 0, 0, nullptr, 0, 0, 0, nullptr, 0, 0, 0, BINS, 1.f);
    launch_gemm(stream, 3, false, MROWS, BINS, 1, FFHB, FFD, 0, 0, L2B, FFD, 0, 0,
                XS, BINS, 0, 0, XS, BINS, 0, 0, nullptr, 0, 0, 0, FFD, 1.f);

    // out[b, 0, h, w] = xs[b, w, h]
    transpose32_kernel<false><<<dim3(BINS / 32, WDIM / 32, BB), tb, 0, stream>>>(
        XS, (long)WDIM * BINS, BINS, outp, (long)WDIM * BINS, WDIM);
}
